// Siglip2Attention_31688268710142
// MI455X (gfx1250) — matrix operand order repaired and checked by reading, not compiled, not run on hardware
//
#include <hip/hip_runtime.h>
#include <hip/hip_bf16.h>

// ---------------------------------------------------------------------------
// Siglip2 attention for MI455X (gfx1250, wave32, WMMA, async-to-LDS staging).
// N=3072 tokens, D=1152, H=16 heads, HD=72 (padded to 96 for K-multiple-of-32),
// 4 equal segments of 768 (block-diagonal attention handled structurally).
// All GEMMs run on v_wmma_f32_16x16x32_f16 with f32 accumulation; GEMM tiles
// are staged with GLOBAL_LOAD_ASYNC_TO_LDS_B128 (ASYNCcnt) double-buffered.
// ---------------------------------------------------------------------------

#define NTOK 3072
#define DMODEL 1152
#define NHEAD 16
#define HDIM 72
#define HDP 96          // padded head dim (3 * 32)
#define SEGLEN 768
#define NSEG 4

typedef __attribute__((ext_vector_type(16))) _Float16 v16h;
typedef __attribute__((ext_vector_type(8)))  _Float16 v8h;
typedef __attribute__((ext_vector_type(8)))  float    v8f;

// ---- CDNA5 async copy: global -> LDS, 16 bytes per lane, ASYNCcnt tracked --
__device__ inline void async_ld16(void* lds_dst, const void* gsrc) {
    asm volatile("global_load_async_to_lds_b128 %0, %1, off"
                 :: "v"((unsigned int)(uintptr_t)lds_dst),       // LDS byte offset (addr[31:0])
                    "v"((unsigned long long)(uintptr_t)gsrc)     // 64-bit global address
                 : "memory");
}
__device__ inline void wait_async0() {
    asm volatile("s_wait_asynccnt 0x0" ::: "memory");
}

// ---- WMMA fragment helpers (layouts per CDNA5 ISA 7.12.2, wave32) ----------
// A (16x32 f16): lane l<16 holds row M=l, K chunks [0..7],[16..23];
//                lane l>=16 holds row M=l-16, K chunks [8..15],[24..31].
__device__ inline v16h load_a_frag(const _Float16* __restrict__ base, int stride, int lane) {
    const int l = lane & 15, half = lane >> 4;
    const _Float16* p0 = base + l * stride + 8 * half;       // K chunk 1 (8 halves)
    const _Float16* p1 = p0 + 16;                            // K chunk 2 (8 halves)
    v8h lo = *(const v8h*)p0;
    v8h hi = *(const v8h*)p1;
    return __builtin_shufflevector(lo, hi, 0,1,2,3,4,5,6,7,8,9,10,11,12,13,14,15);
}

// B (32x16 f16): lane l<16 holds column N=l, K=[0..7],[16..23]; lane>=16: K=[8..15],[24..31].
// Source is B^T row-major: row n = column n, contiguous in K.
__device__ inline v16h load_b_frag(const _Float16* __restrict__ base, int stride, int lane) {
    const int l = lane & 15, half = lane >> 4;
    const _Float16* p = base + l * stride + 8 * half;        // K chunk 1 (8 halves)
    v8h lo = *(const v8h*)p;
    v8h hi = *(const v8h*)(p + 16);
    return __builtin_shufflevector(lo, hi, 0,1,2,3,4,5,6,7,8,9,10,11,12,13,14,15);
}

__device__ inline v8f wmma32(v16h a, v16h b, v8f c) {
    return __builtin_amdgcn_wmma_f32_16x16x32_f16(false, a, false, b, (short)0, c, false, false);
}

// ---------------------------------------------------------------------------
// f32 -> f16 conversion
// ---------------------------------------------------------------------------
__global__ void f32_to_f16_kernel(const float* __restrict__ in, _Float16* __restrict__ out, int n) {
    int i = blockIdx.x * 256 + threadIdx.x;
    if (i < n) out[i] = (_Float16)in[i];
}

// ---------------------------------------------------------------------------
// GEMM: C[M,N] = A[M,K] * B[N,K]^T + bias[N]   (A,B f16, C f32)
// block tile 64x128, 8 waves, each wave -> 16x64 (4 accum tiles), K step 32.
// Double-buffered LDS staging via GLOBAL_LOAD_ASYNC_TO_LDS_B128: prefetch of
// K-tile i+1 overlaps the 4 WMMAs of tile i; one barrier per K-step.
// ---------------------------------------------------------------------------
__global__ __launch_bounds__(256) void gemm_bias_f16_kernel(
    const _Float16* __restrict__ A, const _Float16* __restrict__ B,
    const float* __restrict__ bias, float* __restrict__ C,
    int M, int N, int K)
{
    __shared__ _Float16 As[2][64 * 48];   // 64 rows x 32 halves, stride 48 (16B aligned)
    __shared__ _Float16 Bs[2][128 * 48];

    const int tid  = threadIdx.x;
    const int lane = tid & 31;
    const int wave = tid >> 5;
    const int m_blk  = wave >> 1;       // 0..3
    const int n_half = wave & 1;        // 0..1
    const int m0 = blockIdx.y * 64;
    const int n0 = blockIdx.x * 128;

    // staging assignment: A tile = 64x32 halves (one 16B chunk / thread),
    //                     B tile = 128x32 halves (two 16B chunks / thread)
    const int ra = tid >> 2, ca = (tid & 3) * 8;
    const int rb = tid >> 1, cb = (tid & 1) * 16;
    const _Float16* Asrc = &A[(size_t)(m0 + ra) * K + ca];
    const _Float16* Bsrc = &B[(size_t)(n0 + rb) * K + cb];

    v8f zero = {0.f,0.f,0.f,0.f,0.f,0.f,0.f,0.f};
    v8f acc[4] = {zero, zero, zero, zero};

    // prologue: stage K-tile 0 into buffer 0
    async_ld16(&As[0][ra * 48 + ca], Asrc);
    async_ld16(&Bs[0][rb * 48 + cb], Bsrc);
    async_ld16(&Bs[0][rb * 48 + cb + 8], Bsrc + 8);
    wait_async0();
    __syncthreads();

    const int nsteps = K >> 5;
    for (int i = 0; i < nsteps; ++i) {
        const int buf = i & 1;
        if (i + 1 < nsteps) {
            const int k1 = (i + 1) << 5;
            async_ld16(&As[buf ^ 1][ra * 48 + ca], Asrc + k1);
            async_ld16(&Bs[buf ^ 1][rb * 48 + cb], Bsrc + k1);
            async_ld16(&Bs[buf ^ 1][rb * 48 + cb + 8], Bsrc + k1 + 8);
        }
        // all fragment loads first, then 4 back-to-back WMMAs
        v16h a  = load_a_frag(&As[buf][(m_blk * 16) * 48], 48, lane);
        v16h b0 = load_b_frag(&Bs[buf][(n_half * 64 +  0) * 48], 48, lane);
        v16h b1 = load_b_frag(&Bs[buf][(n_half * 64 + 16) * 48], 48, lane);
        v16h b2 = load_b_frag(&Bs[buf][(n_half * 64 + 32) * 48], 48, lane);
        v16h b3 = load_b_frag(&Bs[buf][(n_half * 64 + 48) * 48], 48, lane);
        acc[0] = wmma32(a, b0, acc[0]);
        acc[1] = wmma32(a, b1, acc[1]);
        acc[2] = wmma32(a, b2, acc[2]);
        acc[3] = wmma32(a, b3, acc[3]);
        wait_async0();          // prefetch into buf^1 complete
        __syncthreads();        // all waves done reading buf / writing buf^1
    }

    // C layout: element (m = jj + 8*half, n = lane&15) in vgpr jj
    const int l = lane & 15, half = lane >> 4;
#pragma unroll
    for (int j = 0; j < 4; ++j) {
        int col = n0 + n_half * 64 + j * 16 + l;
        float bv = bias[col];
#pragma unroll
        for (int jj = 0; jj < 8; ++jj) {
            int row = m0 + m_blk * 16 + jj + 8 * half;
            C[(size_t)row * N + col] = acc[j][jj] + bv;
        }
    }
}

// ---------------------------------------------------------------------------
// RoPE + pack:  q,k -> [H][N][96] f16 (rotated, zero padded 72..95)
//               v   -> [H][96][N] f16 transposed (zero padded rows 72..95)
// ---------------------------------------------------------------------------
__global__ void rope_pack_kernel(
    const float* __restrict__ qf, const float* __restrict__ kf, const float* __restrict__ vf,
    const float* __restrict__ rot,
    _Float16* __restrict__ q16, _Float16* __restrict__ k16, _Float16* __restrict__ vT16)
{
    int idx = blockIdx.x * 256 + threadIdx.x;
    const int total = NTOK * NHEAD * HDP;
    if (idx >= total) return;
    int d = idx % HDP;
    int h = (idx / HDP) & (NHEAD - 1);
    int n = idx / (HDP * NHEAD);

    float qo = 0.f, ko = 0.f, vo = 0.f;
    if (d < HDIM) {
        int src = n * DMODEL + h * HDIM + d;
        int fi  = (d < 36) ? d : (d - 36);
        float fr = rot[n * 36 + fi];
        float c = __cosf(fr), s = __sinf(fr);
        float qv = qf[src], kv = kf[src];
        float qr = (d < 36) ? -qf[src + 36] : qf[src - 36];
        float kr = (d < 36) ? -kf[src + 36] : kf[src - 36];
        qo = qv * c + qr * s;
        ko = kv * c + kr * s;
        vo = vf[src];
    }
    q16[((size_t)h * NTOK + n) * HDP + d] = (_Float16)qo;
    k16[((size_t)h * NTOK + n) * HDP + d] = (_Float16)ko;
    vT16[((size_t)h * HDP + d) * NTOK + n] = (_Float16)vo;
}

// ---------------------------------------------------------------------------
// Attention per (q-tile 64, head, segment 768).
// LDS: scores 64x768 f32 (192KB) + P 64x768 f16 (96KB) + reduce (1KB) -> 289KB
// (fits the 320KB/WGP LDS of CDNA5; dynamic LDS)
// ---------------------------------------------------------------------------
__global__ __launch_bounds__(256) void attn_kernel(
    const _Float16* __restrict__ q16, const _Float16* __restrict__ k16,
    const _Float16* __restrict__ vT16, _Float16* __restrict__ attn16)
{
    extern __shared__ char smem[];
    float*    S   = (float*)smem;                                  // [64][768] f32
    _Float16* P   = (_Float16*)(smem + 64 * 768 * 4);              // [64][768] f16
    float*    red = (float*)(smem + 64 * 768 * 4 + 64 * 768 * 2);  // [64][4]

    const int qt  = blockIdx.x;       // 0..11
    const int h   = blockIdx.y;       // 0..15
    const int sgm = blockIdx.z;       // 0..3
    const int tid = threadIdx.x, lane = tid & 31, wave = tid >> 5;
    const int l = lane & 15, half = lane >> 4;
    const int seg0 = sgm * SEGLEN;
    const int row0 = seg0 + qt * 64;
    const float scale = 0.117851130197758f;   // 1/sqrt(72)

    const _Float16* qh = q16 + (size_t)h * NTOK * HDP;
    const _Float16* kh = k16 + (size_t)h * NTOK * HDP;

    // ---- Phase 1: S = (Q K^T) * scale, K-dim = 96 (3 wmma steps) ----
    {
        const int m_blk = wave >> 1, n_half = wave & 1;
        const _Float16* arow = qh + (size_t)(row0 + m_blk * 16) * HDP;
        v16h a0 = load_a_frag(arow,      HDP, lane);
        v16h a1 = load_a_frag(arow + 32, HDP, lane);
        v16h a2 = load_a_frag(arow + 64, HDP, lane);
        for (int nt = 0; nt < 24; ++nt) {
            int n0 = n_half * 384 + nt * 16;
            const _Float16* kb = kh + (size_t)(seg0 + n0) * HDP;
            v16h b0 = load_b_frag(kb,      HDP, lane);
            v16h b1 = load_b_frag(kb + 32, HDP, lane);
            v16h b2 = load_b_frag(kb + 64, HDP, lane);
            v8f acc = {0.f,0.f,0.f,0.f,0.f,0.f,0.f,0.f};
            acc = wmma32(a0, b0, acc);
            acc = wmma32(a1, b1, acc);
            acc = wmma32(a2, b2, acc);
#pragma unroll
            for (int jj = 0; jj < 8; ++jj) {
                int m = m_blk * 16 + jj + 8 * half;
                S[m * 768 + n0 + l] = acc[jj] * scale;
            }
        }
    }
    __syncthreads();

    // ---- Phase 2: row softmax over 768 keys (4 threads per row) ----
    {
        int row = tid >> 2, part = tid & 3;
        float* Sr = S + row * 768 + part * 192;
        float lm = -3.0e38f;
        for (int i = 0; i < 192; ++i) lm = fmaxf(lm, Sr[i]);
        red[row * 4 + part] = lm;
        __syncthreads();
        float rm = fmaxf(fmaxf(red[row * 4 + 0], red[row * 4 + 1]),
                         fmaxf(red[row * 4 + 2], red[row * 4 + 3]));
        __syncthreads();
        float ls = 0.f;
        for (int i = 0; i < 192; ++i) { float e = __expf(Sr[i] - rm); Sr[i] = e; ls += e; }
        red[row * 4 + part] = ls;
        __syncthreads();
        float inv = 1.0f / (red[row * 4 + 0] + red[row * 4 + 1] +
                            red[row * 4 + 2] + red[row * 4 + 3]);
        _Float16* Pr = P + row * 768 + part * 192;
        for (int i = 0; i < 192; ++i) Pr[i] = (_Float16)(Sr[i] * inv);
    }
    __syncthreads();

    // ---- Phase 3: O = P @ V, K-dim = 768 (24 wmma steps per tile) ----
    {
        const _Float16* vh = vT16 + (size_t)h * HDP * NTOK;
        for (int t = wave; t < 24; t += 8) {
            int m_blk = t / 6, n_blk = t % 6;
            v8f acc = {0.f,0.f,0.f,0.f,0.f,0.f,0.f,0.f};
            for (int kk = 0; kk < SEGLEN; kk += 32) {
                v16h a = load_a_frag(P + (m_blk * 16) * 768 + kk, 768, lane);
                v16h b = load_b_frag(vh + (size_t)(n_blk * 16) * NTOK + seg0 + kk, NTOK, lane);
                acc = wmma32(a, b, acc);
            }
            int col = n_blk * 16 + l;
            if (col < HDIM) {
#pragma unroll
                for (int jj = 0; jj < 8; ++jj) {
                    int row = row0 + m_blk * 16 + jj + 8 * half;
                    attn16[(size_t)row * DMODEL + h * HDIM + col] = (_Float16)acc[jj];
                }
            }
        }
    }
}

// ---------------------------------------------------------------------------
// Host-side launch
// ---------------------------------------------------------------------------
extern "C" void kernel_launch(void* const* d_in, const int* in_sizes, int n_in,
                              void* d_out, int out_size, void* d_ws, size_t ws_size,
                              hipStream_t stream) {
    const float* hs  = (const float*)d_in[0];
    const float* rot = (const float*)d_in[1];
    // d_in[2] = cu_seqlens (equal 768 segments; handled structurally)
    const float* wq = (const float*)d_in[3];
    const float* bq = (const float*)d_in[4];
    const float* wk = (const float*)d_in[5];
    const float* bk = (const float*)d_in[6];
    const float* wv = (const float*)d_in[7];
    const float* bv = (const float*)d_in[8];
    const float* wo = (const float*)d_in[9];
    const float* bo = (const float*)d_in[10];
    float* out = (float*)d_out;

    const size_t ND = (size_t)NTOK * DMODEL;       // 3,538,944
    const size_t DD = (size_t)DMODEL * DMODEL;     // 1,327,104
    const size_t HP = (size_t)NHEAD * NTOK * HDP;  // 4,718,592

    char* ws = (char*)d_ws;
    _Float16* hs16  = (_Float16*)ws;                 ws += ND * 2;
    _Float16* wq16  = (_Float16*)ws;                 ws += DD * 2;
    _Float16* wk16  = (_Float16*)ws;                 ws += DD * 2;
    _Float16* wv16  = (_Float16*)ws;                 ws += DD * 2;
    _Float16* wo16  = (_Float16*)ws;                 ws += DD * 2;
    float*    qf    = (float*)ws;                    ws += ND * 4;
    float*    kf    = (float*)ws;                    ws += ND * 4;
    float*    vf    = (float*)ws;                    ws += ND * 4;
    _Float16* q16   = (_Float16*)ws;                 ws += HP * 2;
    _Float16* k16   = (_Float16*)ws;                 ws += HP * 2;
    _Float16* vT16  = (_Float16*)ws;                 ws += HP * 2;
    _Float16* at16  = (_Float16*)ws;                 ws += ND * 2;

    // 1) f32 -> f16 conversions
    f32_to_f16_kernel<<<(int)((ND + 255) / 256), 256, 0, stream>>>(hs, hs16, (int)ND);
    f32_to_f16_kernel<<<(int)((DD + 255) / 256), 256, 0, stream>>>(wq, wq16, (int)DD);
    f32_to_f16_kernel<<<(int)((DD + 255) / 256), 256, 0, stream>>>(wk, wk16, (int)DD);
    f32_to_f16_kernel<<<(int)((DD + 255) / 256), 256, 0, stream>>>(wv, wv16, (int)DD);
    f32_to_f16_kernel<<<(int)((DD + 255) / 256), 256, 0, stream>>>(wo, wo16, (int)DD);

    // 2) QKV projections (WMMA GEMMs, async-to-LDS staged)
    dim3 gg(DMODEL / 128, NTOK / 64);  // (9, 48)
    gemm_bias_f16_kernel<<<gg, 256, 0, stream>>>(hs16, wq16, bq, qf, NTOK, DMODEL, DMODEL);
    gemm_bias_f16_kernel<<<gg, 256, 0, stream>>>(hs16, wk16, bk, kf, NTOK, DMODEL, DMODEL);
    gemm_bias_f16_kernel<<<gg, 256, 0, stream>>>(hs16, wv16, bv, vf, NTOK, DMODEL, DMODEL);

    // 3) RoPE + pack (q,k per-head padded; v transposed per head)
    const int rtot = NTOK * NHEAD * HDP;
    rope_pack_kernel<<<(rtot + 255) / 256, 256, 0, stream>>>(qf, kf, vf, rot, q16, k16, vT16);

    // 4) Block-diagonal attention: 12 q-tiles x 16 heads x 4 segments
    const size_t smem = (size_t)64 * 768 * 4 + (size_t)64 * 768 * 2 + 64 * 4 * 4; // 295,936 B
    attn_kernel<<<dim3(SEGLEN / 64, NHEAD, NSEG), 256, smem, stream>>>(q16, k16, vT16, at16);

    // 5) Output projection
    gemm_bias_f16_kernel<<<gg, 256, 0, stream>>>(at16, wo16, bo, out, NTOK, DMODEL, DMODEL);
}
